// Clip_34746285424966
// MI455X (gfx1250) — matrix operand order repaired and checked by reading, not compiled, not run on hardware
//
#include <hip/hip_runtime.h>
#include <hip/hip_bf16.h>

typedef __attribute__((ext_vector_type(16))) _Float16 v16h;
typedef __attribute__((ext_vector_type(8)))  _Float16 v8h;
typedef __attribute__((ext_vector_type(2)))  _Float16 v2h;
typedef __attribute__((ext_vector_type(8)))  float    v8f;
typedef __attribute__((ext_vector_type(4)))  float    v4f;

#define LDS_TOTAL 63488
#define OFF_H1    0        // ph1 out: h1 f16 128x136 (34816) ; ph3: logits f32 128x16
#define OFF_EMB   34816    // ph1: emb f32 128x36 (18432)    ; ph2/3: h2 f16 128x72 (18432)
#define OFF_WT    53248    // ph1: W1T f16 128x40 (10240)    ; ph2: W2T 64x40 ; ph3: W3T 16x72

__device__ __forceinline__ v16h cat16(v8h lo, v8h hi) {
    return __builtin_shufflevector(lo, hi, 0,1,2,3,4,5,6,7,8,9,10,11,12,13,14,15);
}

__global__ __launch_bounds__(256)
void clip_mlp_fused(const float* __restrict__ emb,
                    const float* __restrict__ W1, const float* __restrict__ b1,
                    const float* __restrict__ W2, const float* __restrict__ b2,
                    const float* __restrict__ W3, const float* __restrict__ b3,
                    float* __restrict__ out)
{
    constexpr int H = 16, E = 768, D1 = 128, D2 = 64, C = 10;
    __shared__ __align__(16) unsigned char smem[LDS_TOTAL];
    _Float16* ldsH1  = (_Float16*)(smem + OFF_H1);    // stride 136
    float*    ldsEmb = (float*)   (smem + OFF_EMB);   // stride 36 (f32)
    _Float16* ldsBT  = (_Float16*)(smem + OFF_WT);    // stride 40
    _Float16* ldsW2T = (_Float16*)(smem + OFF_WT);    // stride 40
    _Float16* ldsH2  = (_Float16*)(smem + OFF_EMB);   // stride 72
    _Float16* ldsW3T = (_Float16*)(smem + OFF_WT);    // stride 72
    float*    ldsLg  = (float*)   (smem + OFF_H1);    // 128 x 16 f32

    const int t    = threadIdx.x;
    const int wid  = t >> 5;
    const int lane = t & 31;
    const int l15  = lane & 15;
    const int hsel = lane >> 4;       // 0 / 1
    const int abase = hsel << 3;      // A-frag K base: 0 / 8
    const int bbase = hsel << 3;      // B-frag K base: 0 / 8
    const int h    = blockIdx.y;
    const int row0 = blockIdx.x * 128;
    const int mrow = (wid << 4) + l15;     // this lane's A-row (0..127)

    const float* W1h = W1 + (size_t)h * E  * D1;
    const float* W2h = W2 + (size_t)h * D1 * D2;
    const float* W3h = W3 + (size_t)h * D2 * C;

    const v8f zacc = {0.f,0.f,0.f,0.f,0.f,0.f,0.f,0.f};

    // ---------------- GEMM1: h1 = relu(emb(128x768) @ W1[h](768x128) + b1) ----
    v8f acc[8];
#pragma unroll
    for (int n = 0; n < 8; ++n) acc[n] = zacc;

    // emb async staging map: thread t copies row rA, 16 floats (4x B128)
    const int rA = t >> 1;                 // 0..127
    const int cA = (t & 1) << 4;           // 0 / 16 (floats)
    const unsigned ldsbase = (unsigned)(uintptr_t)smem;
    const unsigned embLds  = ldsbase + OFF_EMB + (unsigned)(rA * 36 + cA) * 4u;
    const unsigned embGoff = (unsigned)(rA * E + cA) * 4u;   // byte offset vs saddr

    // W-transpose staging map: thread t handles k-pair (kk2,kk2+1) x 8 d's
    const int kk2 = (t & 15) << 1;         // 0..30 step 2
    const int dW  = (t >> 4) << 3;         // 0..120 step 8

    for (int kb = 0; kb < E / 32; ++kb) {
        // ---- async-copy emb chunk (raw f32) into LDS [128][36] ----
        {
            const float* sbase = emb + (size_t)row0 * E + kb * 32;  // uniform -> SGPRs
            asm volatile("global_load_async_to_lds_b128 %0, %1, %2 offset:0"
                         :: "v"(embLds), "v"(embGoff), "s"(sbase) : "memory");
            asm volatile("global_load_async_to_lds_b128 %0, %1, %2 offset:16"
                         :: "v"(embLds), "v"(embGoff), "s"(sbase) : "memory");
            asm volatile("global_load_async_to_lds_b128 %0, %1, %2 offset:32"
                         :: "v"(embLds), "v"(embGoff), "s"(sbase) : "memory");
            asm volatile("global_load_async_to_lds_b128 %0, %1, %2 offset:48"
                         :: "v"(embLds), "v"(embGoff), "s"(sbase) : "memory");
        }
        // ---- stage W1 chunk transposed -> [d:128][k:40] f16, packed b32 stores ----
        {
            const float* ws0 = W1h + (size_t)(kb * 32 + kk2) * D1 + dW;
            const float* ws1 = ws0 + D1;
            if (kb + 1 < E / 32)
                __builtin_prefetch(ws0 + 32 * D1, 0, 1);
            v4f r0a = *(const v4f*)(ws0);
            v4f r0b = *(const v4f*)(ws0 + 4);
            v4f r1a = *(const v4f*)(ws1);
            v4f r1b = *(const v4f*)(ws1 + 4);
#pragma unroll
            for (int i = 0; i < 4; ++i) {
                v2h p = { (_Float16)r0a[i], (_Float16)r1a[i] };
                *(v2h*)(ldsBT + (dW + i) * 40 + kk2) = p;
            }
#pragma unroll
            for (int i = 0; i < 4; ++i) {
                v2h p = { (_Float16)r0b[i], (_Float16)r1b[i] };
                *(v2h*)(ldsBT + (dW + 4 + i) * 40 + kk2) = p;
            }
        }
        asm volatile("s_wait_asynccnt 0x0" ::: "memory");
        __syncthreads();

        // ---- A fragment: read f32 runs, convert to f16 ----
        const float* ar = ldsEmb + mrow * 36;
        v4f a0 = *(const v4f*)(ar + abase);
        v4f a1 = *(const v4f*)(ar + abase + 4);
        v4f a2 = *(const v4f*)(ar + abase + 16);
        v4f a3 = *(const v4f*)(ar + abase + 20);
        v16h a = { (_Float16)a0[0],(_Float16)a0[1],(_Float16)a0[2],(_Float16)a0[3],
                   (_Float16)a1[0],(_Float16)a1[1],(_Float16)a1[2],(_Float16)a1[3],
                   (_Float16)a2[0],(_Float16)a2[1],(_Float16)a2[2],(_Float16)a2[3],
                   (_Float16)a3[0],(_Float16)a3[1],(_Float16)a3[2],(_Float16)a3[3] };
#pragma unroll
        for (int n = 0; n < 8; ++n) {
            const _Float16* bc = ldsBT + (n * 16 + l15) * 40 + bbase;
            v16h bm = cat16(*(const v8h*)bc, *(const v8h*)(bc + 16));
            acc[n] = __builtin_amdgcn_wmma_f32_16x16x32_f16(
                false, a, false, bm, (short)0, acc[n], false, false);
        }
        __syncthreads();
    }

    // epilogue: bias + relu -> ldsH1 (f16, stride 136)
#pragma unroll
    for (int n = 0; n < 8; ++n) {
        const int N = n * 16 + l15;
        const float bv = b1[h * D1 + N];
#pragma unroll
        for (int v = 0; v < 8; ++v) {
            const int M = (wid << 4) + v + abase;
            float val = acc[n][v] + bv;
            ldsH1[M * 136 + N] = (_Float16)fmaxf(val, 0.f);
        }
    }
    __syncthreads();

    // ---------------- GEMM2: h2 = relu(h1(128x128) @ W2[h](128x64) + b2) -----
    v8f acc2[4];
#pragma unroll
    for (int n = 0; n < 4; ++n) acc2[n] = zacc;

    const int n0W2 = (t >> 4) << 2;   // 0..60 step 4
    for (int kb = 0; kb < 4; ++kb) {
        {
            const float* ws0 = W2h + (size_t)(kb * 32 + kk2) * D2 + n0W2;
            const float* ws1 = ws0 + D2;
            v4f r0 = *(const v4f*)ws0;
            v4f r1 = *(const v4f*)ws1;
#pragma unroll
            for (int i = 0; i < 4; ++i) {
                v2h p = { (_Float16)r0[i], (_Float16)r1[i] };
                *(v2h*)(ldsW2T + (n0W2 + i) * 40 + kk2) = p;
            }
        }
        __syncthreads();

        const _Float16* ar = ldsH1 + mrow * 136 + kb * 32;
        v16h a = cat16(*(const v8h*)(ar + abase), *(const v8h*)(ar + abase + 16));
#pragma unroll
        for (int n = 0; n < 4; ++n) {
            const _Float16* bc = ldsW2T + (n * 16 + l15) * 40 + bbase;
            v16h bm = cat16(*(const v8h*)bc, *(const v8h*)(bc + 16));
            acc2[n] = __builtin_amdgcn_wmma_f32_16x16x32_f16(
                false, a, false, bm, (short)0, acc2[n], false, false);
        }
        __syncthreads();
    }

#pragma unroll
    for (int n = 0; n < 4; ++n) {
        const int N = n * 16 + l15;
        const float bv = b2[h * D2 + N];
#pragma unroll
        for (int v = 0; v < 8; ++v) {
            const int M = (wid << 4) + v + abase;
            float val = acc2[n][v] + bv;
            ldsH2[M * 72 + N] = (_Float16)fmaxf(val, 0.f);
        }
    }
    __syncthreads();

    // ---------------- GEMM3: logits = h2(128x64) @ W3[h](64x10->16) + b3 -----
    {   // stage W3 transposed & zero-padded to 16 cols: [c:16][k:72]
        const int k  = t >> 2;
        const int c0 = (t & 3) << 2;
        const float* ws = W3h + k * C;
#pragma unroll
        for (int i = 0; i < 4; ++i) {
            const int c = c0 + i;
            float w = (c < C) ? ws[c] : 0.f;
            ldsW3T[c * 72 + k] = (_Float16)w;
        }
    }
    __syncthreads();

    v8f acc3 = zacc;
#pragma unroll
    for (int kb = 0; kb < 2; ++kb) {
        const _Float16* ar = ldsH2 + mrow * 72 + kb * 32;
        v16h a = cat16(*(const v8h*)(ar + abase), *(const v8h*)(ar + abase + 16));
        const _Float16* bc = ldsW3T + l15 * 72 + kb * 32 + bbase;
        v16h bm = cat16(*(const v8h*)bc, *(const v8h*)(bc + 16));
        acc3 = __builtin_amdgcn_wmma_f32_16x16x32_f16(
            false, a, false, bm, (short)0, acc3, false, false);
    }

    {   // logits (+bias) -> LDS f32 [128][16]
        const int N = l15;
        const float bv = (N < C) ? b3[h * C + N] : 0.f;
#pragma unroll
        for (int v = 0; v < 8; ++v) {
            const int M = (wid << 4) + v + abase;
            ldsLg[M * 16 + N] = acc3[v] + bv;
        }
    }
    __syncthreads();

    // ---------------- softmax over C=10, one thread per row -------------------
    if (t < 128) {
        const float* lr = ldsLg + t * 16;
        float m = lr[0];
#pragma unroll
        for (int c = 1; c < C; ++c) m = fmaxf(m, lr[c]);
        float e[C];
        float s = 0.f;
#pragma unroll
        for (int c = 0; c < C; ++c) { e[c] = __expf(lr[c] - m); s += e[c]; }
        const float inv = 1.f / s;
        float* op = out + ((size_t)(row0 + t) * H + h) * C;
#pragma unroll
        for (int c = 0; c < C; ++c) op[c] = e[c] * inv;
    }
}

extern "C" void kernel_launch(void* const* d_in, const int* in_sizes, int n_in,
                              void* d_out, int out_size, void* d_ws, size_t ws_size,
                              hipStream_t stream)
{
    (void)in_sizes; (void)n_in; (void)out_size; (void)d_ws; (void)ws_size;
    // inputs: 0:x(int,unused) 1:clip_embedding 2:W1 3:b1 4:W2 5:b2 6:W3 7:b3
    const float* emb = (const float*)d_in[1];
    const float* W1  = (const float*)d_in[2];
    const float* b1  = (const float*)d_in[3];
    const float* W2  = (const float*)d_in[4];
    const float* b2  = (const float*)d_in[5];
    const float* W3  = (const float*)d_in[6];
    const float* b3  = (const float*)d_in[7];
    float* out = (float*)d_out;

    dim3 grid(32768 / 128, 16);   // (row tiles, heads)
    dim3 block(256);              // 8 x wave32
    clip_mlp_fused<<<grid, block, 0, stream>>>(emb, W1, b1, W2, b2, W3, b3, out);
}
